// MultiHeadAttention_5497558138925
// MI455X (gfx1250) — compile-verified, matrix operand order checked by reading, not run on hardware
//
#include <hip/hip_runtime.h>
#include <hip/hip_bf16.h>

// MHA forward for MI455X (gfx1250).
// - all four matmuls on v_wmma_f32_16x16x32_bf16
// - fp32 operands pre-converted to bf16 (weights transposed -> contiguous frags)
// - workgroup tiles staged in LDS by the Tensor Data Mover (tensor_load_to_lds,
//   TENSORcnt), double-buffered so DMA overlaps WMMA
//
// Workspace (bf16 elems): xb 4M | WqkvT 3M | WoutT 1M | q 4M | k 4M | vT 4M |
// ctx 4M  => 24M bf16 = 48 MB of d_ws.

#define BB 2
#define TT 2048
#define CC 1024
#define HH 16
#define HD 64

typedef __attribute__((ext_vector_type(16))) __bf16 v16bf;
typedef __attribute__((ext_vector_type(4)))  __bf16 v4bf;
typedef __attribute__((ext_vector_type(8)))  float  v8f;
typedef __attribute__((ext_vector_type(4)))  unsigned int u32x4;
typedef __attribute__((ext_vector_type(8)))  int i32x8;
typedef __attribute__((ext_vector_type(4)))  int i32x4;

__device__ __forceinline__ v8f wmma_bf16(v16bf a, v16bf b, v8f c) {
  return __builtin_amdgcn_wmma_f32_16x16x32_bf16(false, a, false, b, (short)0, c,
                                                 false, false);
}

// Build a 16x(bf16) fragment from two 16-byte chunks (global or LDS).
__device__ __forceinline__ v16bf load_frag(const __bf16* p0, const __bf16* p1) {
  union { v16bf v; uint4 q[2]; } u;
  u.q[0] = *(const uint4*)p0;
  u.q[1] = *(const uint4*)p1;
  return u.v;
}

// ---------------------------------------------------------------------------
// TDM: async 2D tile load global -> LDS (bf16 elements, data_size=2B).
// Descriptor per CDNA5 ISA ch.8: group0 {count=1, lds_addr, global_addr,
// type=2}; group1 {data_size=1, tensor_dim0/1, tile_dim0/1, dim0 stride}.
// ---------------------------------------------------------------------------
__device__ __forceinline__ void tdm_load_2d(unsigned lds_off, const void* gaddr,
                                            unsigned tensor_d0, unsigned tensor_d1,
                                            unsigned stride0_elems,
                                            unsigned tile_d0, unsigned tile_d1) {
  const unsigned long long ga = (unsigned long long)gaddr;
  u32x4 g0;
  g0[0] = 1u;                                            // count=1 (user D#)
  g0[1] = lds_off;                                       // lds_addr (bytes)
  g0[2] = (unsigned)(ga & 0xffffffffu);
  g0[3] = (unsigned)((ga >> 32) & 0x01ffffffu) | (2u << 30);  // type=2
  i32x8 g1;
  g1[0] = (int)(1u << 16);                               // data_size=1 -> 2B
  g1[1] = (int)((tensor_d0 & 0xffffu) << 16);            // tensor_dim0[15:0]
  g1[2] = (int)((tensor_d0 >> 16) | ((tensor_d1 & 0xffffu) << 16));
  g1[3] = (int)((tensor_d1 >> 16) | (tile_d0 << 16));    // tile_dim0
  g1[4] = (int)(tile_d1 & 0xffffu);                      // tile_dim1, tile_dim2=0
  g1[5] = (int)stride0_elems;                            // tensor_dim0_stride lo
  g1[6] = 0;                                             // stride hi, dim1_stride lo
  g1[7] = 0;
  const i32x4 z4 = {0, 0, 0, 0};
#if __clang_major__ >= 23
  const i32x8 z8 = {0, 0, 0, 0, 0, 0, 0, 0};
  __builtin_amdgcn_tensor_load_to_lds(g0, g1, z4, z4, z8, 0);
#else
  __builtin_amdgcn_tensor_load_to_lds(g0, g1, z4, z4, 0);
#endif
}

__device__ __forceinline__ unsigned lds_off_of(const void* p) {
  return (unsigned)(unsigned long long)p;   // flat LDS addr truncates to offset
}

// ---------------------------------------------------------------------------
// Kernel 0a: straight fp32 -> bf16 convert (x). 4 elems/thread.
// ---------------------------------------------------------------------------
__global__ __launch_bounds__(256) void convert_kernel(
    const float* __restrict__ src, __bf16* __restrict__ dst, int n) {
  const int i = (blockIdx.x * 256 + threadIdx.x) * 4;
  if (i < n) {
    const float4 f = *(const float4*)(src + i);
    v4bf o;
    o.x = (__bf16)f.x; o.y = (__bf16)f.y; o.z = (__bf16)f.z; o.w = (__bf16)f.w;
    *(v4bf*)(dst + i) = o;
  }
}

// ---------------------------------------------------------------------------
// Kernel 0b: tiled transpose + fp32 -> bf16. dst[n][k] = (bf16)src[k][n].
// ---------------------------------------------------------------------------
__global__ __launch_bounds__(256) void transpose_convert_kernel(
    const float* __restrict__ src, __bf16* __restrict__ dst, int rows, int cols) {
  __shared__ float tile[32][33];
  const int bx = blockIdx.x * 32, by = blockIdx.y * 32;
  const int tx = threadIdx.x, ty = threadIdx.y;
#pragma unroll
  for (int i = 0; i < 32; i += 8)
    tile[ty + i][tx] = src[(size_t)(by + ty + i) * cols + bx + tx];
  __syncthreads();
#pragma unroll
  for (int i = 0; i < 32; i += 8)
    dst[(size_t)(bx + ty + i) * rows + by + tx] = (__bf16)tile[tx][ty + i];
}

// ---------------------------------------------------------------------------
// Kernel 1: qkv = x @ Wqkv with fused RoPE; v stored transposed.
// Block (4 waves) computes 16(M) x 256(N); TDM double-buffers A(16x32) and
// B(256x32) tiles in LDS. grid (256, 12), block 128.
// ---------------------------------------------------------------------------
__global__ __launch_bounds__(128) void qkv_rope_kernel(
    const __bf16* __restrict__ xb, const __bf16* __restrict__ WqkvT,
    __bf16* __restrict__ q, __bf16* __restrict__ k, __bf16* __restrict__ vT) {
  __shared__ __attribute__((aligned(16))) __bf16 Atile[2][16 * 32];    //  2 KB
  __shared__ __attribute__((aligned(16))) __bf16 Btile[2][256 * 32];   // 32 KB

  const int lane  = threadIdx.x & 31;
  const int wave  = threadIdx.x >> 5;
  const int strip = blockIdx.y * 4 + wave;        // 0..47
  const int part  = strip >> 4;                    // 0=q 1=k 2=v
  const int head  = strip & 15;
  const int mBase = blockIdx.x * 16;
  const int nBase = blockIdx.y * 256;              // block's B-row window
  const int lrow  = lane & 15;
  const int chunk = lane >> 4;

  v8f acc[4];
#pragma unroll
  for (int g = 0; g < 4; ++g)
#pragma unroll
    for (int i = 0; i < 8; ++i) acc[g][i] = 0.0f;

  const __bf16* Ag = xb + (size_t)mBase * CC;
  const __bf16* Bg = WqkvT + (size_t)nBase * CC;

  if (wave == 0) {
    tdm_load_2d(lds_off_of(&Atile[0][0]), Ag, CC, BB * TT, CC, 32, 16);
    tdm_load_2d(lds_off_of(&Btile[0][0]), Bg, CC, 3 * CC, CC, 32, 256);
  }

  const int NIT = CC / 32;
  for (int it = 0; it < NIT; ++it) {
    const int buf = it & 1;
    if (wave == 0) {
      if (it + 1 < NIT) {
        const int k1 = (it + 1) * 32;
        tdm_load_2d(lds_off_of(&Atile[buf ^ 1][0]), Ag + k1, CC, BB * TT, CC, 32, 16);
        tdm_load_2d(lds_off_of(&Btile[buf ^ 1][0]), Bg + k1, CC, 3 * CC, CC, 32, 256);
        __builtin_amdgcn_s_wait_tensorcnt(2);
      } else {
        __builtin_amdgcn_s_wait_tensorcnt(0);
      }
    }
    __syncthreads();                               // tile[buf] ready for all
    const __bf16* ab = &Atile[buf][0] + lrow * 32 + chunk * 8;
    const v16bf a = load_frag(ab, ab + 16);
#pragma unroll
    for (int g = 0; g < 4; ++g) {
      const __bf16* bb = &Btile[buf][0] + (wave * 64 + g * 16 + lrow) * 32 + chunk * 8;
      acc[g] = wmma_bf16(a, load_frag(bb, bb + 16), acc[g]);
    }
    __syncthreads();                               // done reading tile[buf]
  }

#pragma unroll
  for (int r = 0; r < 8; ++r) {
    const int row  = mBase + r + 8 * chunk;
    const int bIdx = row / TT;
    const int t    = row - bIdx * TT;
    if (part == 2) {
#pragma unroll
      for (int g = 0; g < 4; ++g) {
        const int d = g * 16 + lrow;
        vT[(((size_t)bIdx * HH + head) * HD + d) * TT + t] = (__bf16)acc[g][r];
      }
    } else {
      __bf16* dst = (part == 0) ? q : k;
#pragma unroll
      for (int g = 0; g < 4; ++g) {
        const int d    = g * 16 + lrow;
        const float xv = acc[g][r];
        const float pv = acc[(g + 2) & 3][r];      // RoPE partner (d +/- 32)
        const int d32  = d & 31;
        const float inv = __expf((float)(2 * d32) * (-9.210340371976184f / 64.0f));
        float sn, cs;
        __sincosf((float)t * inv, &sn, &cs);
        const float outv = (d < 32) ? (xv * cs - pv * sn) : (xv * cs + pv * sn);
        dst[(((size_t)bIdx * HH + head) * TT + t) * HD + d] = (__bf16)outv;
      }
    }
  }
}

// ---------------------------------------------------------------------------
// Kernel 2: flash attention. 2 waves share TDM-staged K(32x64) / V(64x32)
// tiles, double-buffered; per-wave P staging. grid (64, 32), block 64.
// ---------------------------------------------------------------------------
__global__ __launch_bounds__(64) void flash_attn_kernel(
    const __bf16* __restrict__ q, const __bf16* __restrict__ k,
    const __bf16* __restrict__ vT, __bf16* __restrict__ ctx) {
  __shared__ __attribute__((aligned(16))) __bf16 Ktile[2][32 * 64];   // 8 KB
  __shared__ __attribute__((aligned(16))) __bf16 Vtile[2][64 * 32];   // 8 KB
  __shared__ __attribute__((aligned(16))) __bf16 plds[2][16 * 32];    // 2 KB

  const int lane  = threadIdx.x & 31;
  const int wave  = threadIdx.x >> 5;
  const int bh    = blockIdx.y;
  const int bIdx  = bh >> 4, head = bh & 15;
  const int q0    = (blockIdx.x * 2 + wave) * 16;
  const int lrow  = lane & 15;
  const int chunk = lane >> 4;

  const __bf16* Qb = q  + (size_t)bh * TT * HD;
  const __bf16* Kb = k  + (size_t)bh * TT * HD;
  const __bf16* Vb = vT + (size_t)bh * HD * TT;
  __bf16* pbase    = &plds[wave][0];

  const __bf16* qp = Qb + (size_t)(q0 + lrow) * HD + chunk * 8;
  const v16bf a0 = load_frag(qp, qp + 16);
  const v16bf a1 = load_frag(qp + 32, qp + 48);

  v8f o[4];
  float m_[8], l_[8];
#pragma unroll
  for (int g = 0; g < 4; ++g)
#pragma unroll
    for (int i = 0; i < 8; ++i) o[g][i] = 0.0f;
#pragma unroll
  for (int r = 0; r < 8; ++r) { m_[r] = -3.0e38f; l_[r] = 0.0f; }

  if (wave == 0) {
    tdm_load_2d(lds_off_of(&Ktile[0][0]), Kb, HD, TT, HD, 64, 32);
    tdm_load_2d(lds_off_of(&Vtile[0][0]), Vb, TT, HD, TT, 32, 64);
  }

  const int NIT = TT / 32;
  for (int it = 0; it < NIT; ++it) {
    const int buf = it & 1;
    if (wave == 0) {
      if (it + 1 < NIT) {
        const int kt1 = (it + 1) * 32;
        tdm_load_2d(lds_off_of(&Ktile[buf ^ 1][0]), Kb + (size_t)kt1 * HD, HD, TT, HD, 64, 32);
        tdm_load_2d(lds_off_of(&Vtile[buf ^ 1][0]), Vb + kt1, TT, HD, TT, 32, 64);
        __builtin_amdgcn_s_wait_tensorcnt(2);
      } else {
        __builtin_amdgcn_s_wait_tensorcnt(0);
      }
    }
    __syncthreads();

    // S = Q @ K^T from LDS tile
    v8f s01[2];
#pragma unroll
    for (int ct = 0; ct < 2; ++ct) {
      const __bf16* kp = &Ktile[buf][0] + (ct * 16 + lrow) * 64 + chunk * 8;
      const v16bf b0 = load_frag(kp, kp + 16);
      const v16bf b1 = load_frag(kp + 32, kp + 48);
      v8f z;
#pragma unroll
      for (int i = 0; i < 8; ++i) z[i] = 0.0f;
      s01[ct] = wmma_bf16(a1, b1, wmma_bf16(a0, b0, z));
    }

    // Online softmax; stage P (bf16) in wave-private LDS
#pragma unroll
    for (int r = 0; r < 8; ++r) {
      const float v0 = s01[0][r] * 0.125f;
      const float v1 = s01[1][r] * 0.125f;
      float mx = fmaxf(v0, v1);
#pragma unroll
      for (int msk = 1; msk < 16; msk <<= 1) mx = fmaxf(mx, __shfl_xor(mx, msk, 32));
      const float nm = fmaxf(m_[r], mx);
      const float p0 = __expf(v0 - nm);
      const float p1 = __expf(v1 - nm);
      float rs = p0 + p1;
#pragma unroll
      for (int msk = 1; msk < 16; msk <<= 1) rs += __shfl_xor(rs, msk, 32);
      const float alpha = __expf(m_[r] - nm);
      m_[r] = nm;
      l_[r] = l_[r] * alpha + rs;
#pragma unroll
      for (int g = 0; g < 4; ++g) o[g][r] *= alpha;
      const int row = r + 8 * chunk;
      pbase[row * 32 + lrow]      = (__bf16)p0;
      pbase[row * 32 + lrow + 16] = (__bf16)p1;
    }

    const __bf16* pr = pbase + lrow * 32 + chunk * 8;
    const v16bf pa = load_frag(pr, pr + 16);

    // O += P @ V from LDS tile
#pragma unroll
    for (int g = 0; g < 4; ++g) {
      const __bf16* vp = &Vtile[buf][0] + (g * 16 + lrow) * 32 + chunk * 8;
      o[g] = wmma_bf16(pa, load_frag(vp, vp + 16), o[g]);
    }
    __syncthreads();
  }

#pragma unroll
  for (int r = 0; r < 8; ++r) {
    const float inv  = 1.0f / l_[r];
    const int   trow = q0 + r + 8 * chunk;
#pragma unroll
    for (int g = 0; g < 4; ++g) {
      const int c = head * HD + g * 16 + lrow;
      ctx[((size_t)bIdx * TT + trow) * CC + c] = (__bf16)(o[g][r] * inv);
    }
  }
}

// ---------------------------------------------------------------------------
// Kernel 3: out = ctx @ Wout + bout, TDM-staged tiles. grid (256, 4), block 128.
// ---------------------------------------------------------------------------
__global__ __launch_bounds__(128) void out_proj_kernel(
    const __bf16* __restrict__ ctx, const __bf16* __restrict__ WoutT,
    const float* __restrict__ bout, float* __restrict__ out) {
  __shared__ __attribute__((aligned(16))) __bf16 Atile[2][16 * 32];
  __shared__ __attribute__((aligned(16))) __bf16 Btile[2][256 * 32];

  const int lane  = threadIdx.x & 31;
  const int wave  = threadIdx.x >> 5;
  const int mBase = blockIdx.x * 16;
  const int nBase = blockIdx.y * 256;
  const int lrow  = lane & 15;
  const int chunk = lane >> 4;

  v8f acc[4];
#pragma unroll
  for (int g = 0; g < 4; ++g)
#pragma unroll
    for (int i = 0; i < 8; ++i) acc[g][i] = 0.0f;

  const __bf16* Ag = ctx + (size_t)mBase * CC;
  const __bf16* Bg = WoutT + (size_t)nBase * CC;

  if (wave == 0) {
    tdm_load_2d(lds_off_of(&Atile[0][0]), Ag, CC, BB * TT, CC, 32, 16);
    tdm_load_2d(lds_off_of(&Btile[0][0]), Bg, CC, CC, CC, 32, 256);
  }

  const int NIT = CC / 32;
  for (int it = 0; it < NIT; ++it) {
    const int buf = it & 1;
    if (wave == 0) {
      if (it + 1 < NIT) {
        const int k1 = (it + 1) * 32;
        tdm_load_2d(lds_off_of(&Atile[buf ^ 1][0]), Ag + k1, CC, BB * TT, CC, 32, 16);
        tdm_load_2d(lds_off_of(&Btile[buf ^ 1][0]), Bg + k1, CC, CC, CC, 32, 256);
        __builtin_amdgcn_s_wait_tensorcnt(2);
      } else {
        __builtin_amdgcn_s_wait_tensorcnt(0);
      }
    }
    __syncthreads();
    const __bf16* ab = &Atile[buf][0] + lrow * 32 + chunk * 8;
    const v16bf a = load_frag(ab, ab + 16);
#pragma unroll
    for (int g = 0; g < 4; ++g) {
      const __bf16* bb = &Btile[buf][0] + (wave * 64 + g * 16 + lrow) * 32 + chunk * 8;
      acc[g] = wmma_bf16(a, load_frag(bb, bb + 16), acc[g]);
    }
    __syncthreads();
  }

#pragma unroll
  for (int r = 0; r < 8; ++r) {
    const int row = mBase + r + 8 * chunk;
#pragma unroll
    for (int g = 0; g < 4; ++g) {
      const int col = nBase + wave * 64 + g * 16 + lrow;
      out[(size_t)row * CC + col] = acc[g][r] + bout[col];
    }
  }
}

// ---------------------------------------------------------------------------
extern "C" void kernel_launch(void* const* d_in, const int* in_sizes, int n_in,
                              void* d_out, int out_size, void* d_ws, size_t ws_size,
                              hipStream_t stream) {
  const float* x    = (const float*)d_in[0];
  const float* Wqkv = (const float*)d_in[1];
  const float* Wout = (const float*)d_in[2];
  const float* bout = (const float*)d_in[3];
  float* out        = (float*)d_out;

  const size_t M = (size_t)BB * TT;                 // 4096
  __bf16* xb    = (__bf16*)d_ws;                    // 4M
  __bf16* WqkvT = xb + M * CC;                      // 3M
  __bf16* WoutT = WqkvT + (size_t)3 * CC * CC;      // 1M
  __bf16* qb    = WoutT + (size_t)CC * CC;          // 4M
  __bf16* kb    = qb + (size_t)BB * HH * TT * HD;   // 4M
  __bf16* vTb   = kb + (size_t)BB * HH * TT * HD;   // 4M
  __bf16* ctx   = vTb + (size_t)BB * HH * TT * HD;  // 4M

  convert_kernel<<<dim3((int)(M * CC / 4 / 256)), 256, 0, stream>>>(x, xb, (int)(M * CC));
  transpose_convert_kernel<<<dim3(3 * CC / 32, CC / 32), dim3(32, 8), 0, stream>>>(
      Wqkv, WqkvT, CC, 3 * CC);
  transpose_convert_kernel<<<dim3(CC / 32, CC / 32), dim3(32, 8), 0, stream>>>(
      Wout, WoutT, CC, CC);

  qkv_rope_kernel<<<dim3(256, 12), 128, 0, stream>>>(xb, WqkvT, qb, kb, vTb);
  flash_attn_kernel<<<dim3(64, 32), 64, 0, stream>>>(qb, kb, vTb, ctx);
  out_proj_kernel<<<dim3(256, 4), 128, 0, stream>>>(ctx, WoutT, bout, out);
}
